// SubGraphTransformerEncoderLayer_24215025614959
// MI455X (gfx1250) — compile-verified, matrix operand order checked by reading, not run on hardware
//
#include <hip/hip_runtime.h>
#include <hip/hip_bf16.h>

typedef __attribute__((ext_vector_type(16))) __bf16 v16bf;
typedef __attribute__((ext_vector_type(8)))  __bf16 v8bf;
typedef __attribute__((ext_vector_type(8)))  float  v8f;
typedef __attribute__((ext_vector_type(4)))  float  fvec4;

static __device__ __forceinline__ v8f zero8() {
    v8f r;
#pragma unroll
    for (int i = 0; i < 8; i++) r[i] = 0.f;
    return r;
}

// Load a 16x32 bf16 fragment in the CDNA5 WMMA A-matrix layout from a
// row-major matrix (row stride `ld` elements), tile origin (row0, k0).
// Lane l holds row (l&15); element j maps to k = base + j + 8*(j>>3),
// base = (l>>4)*8  -> two contiguous 16-byte loads per lane.
// The same function loads B fragments when `p` is the row-major transpose
// of the B operand (rows indexed by N).
static __device__ __forceinline__ v16bf load_frag(const __bf16* p, int ld, int row0, int k0, int lane) {
    const int m    = lane & 15;
    const int base = (lane >> 4) << 3;
    const __bf16* s = p + (size_t)(row0 + m) * ld + (k0 + base);
    const v8bf lo = *(const v8bf*)s;         // k = base .. base+7
    const v8bf hi = *(const v8bf*)(s + 16);  // k = base+16 .. base+23
    v16bf r;
#pragma unroll
    for (int i = 0; i < 8; i++) { r[i] = lo[i]; r[i + 8] = hi[i]; }
    return r;
}

// ---------------------------------------------------------------- weights
// Wt[n][k] = (bf16) W[k][n]   (W is K x N row-major)
__global__ void transpose_f32_to_bf16_kernel(const float* __restrict__ W, __bf16* __restrict__ Wt,
                                             int K, int N) {
    int idx = blockIdx.x * blockDim.x + threadIdx.x;
    if (idx >= K * N) return;
    int n = idx / K, k = idx % K;
    Wt[idx] = (__bf16)W[(size_t)k * N + n];
}

// bf16 tiled transpose: out[C][R] = in[R][C]
__global__ void transpose_bf16_kernel(const __bf16* __restrict__ in, __bf16* __restrict__ out,
                                      int R, int C) {
    __shared__ __bf16 tile[32][33];
    const int tc = blockIdx.x % (C / 32);
    const int tr = blockIdx.x / (C / 32);
    const int x = threadIdx.x, y = threadIdx.y;  // 32 x 8
#pragma unroll
    for (int i = 0; i < 32; i += 8)
        tile[y + i][x] = in[(size_t)(tr * 32 + y + i) * C + tc * 32 + x];
    __syncthreads();
#pragma unroll
    for (int i = 0; i < 32; i += 8)
        out[(size_t)(tc * 32 + y + i) * R + tr * 32 + x] = tile[x][y + i];
}

// ---------------------------------------------------------------- layernorm
// One wave per row of 256; each lane owns 8 elements; shfl-tree reductions.
__global__ void layernorm_bf16_kernel(const float* __restrict__ X, const float* __restrict__ g,
                                      const float* __restrict__ b, __bf16* __restrict__ out, int rows) {
    const int gid = blockIdx.x * blockDim.x + threadIdx.x;
    const int wave = gid >> 5, lane = gid & 31;
    if (wave >= rows) return;
    const float4* r = (const float4*)(X + (size_t)wave * 256);
    const float4 a0 = r[lane], a1 = r[lane + 32];
    float s = a0.x + a0.y + a0.z + a0.w + a1.x + a1.y + a1.z + a1.w;
#pragma unroll
    for (int d = 1; d < 32; d <<= 1) s += __shfl_xor(s, d, 32);
    const float mu = s * (1.f / 256.f);
    const float d0 = a0.x - mu, d1 = a0.y - mu, d2 = a0.z - mu, d3 = a0.w - mu;
    const float d4 = a1.x - mu, d5 = a1.y - mu, d6 = a1.z - mu, d7 = a1.w - mu;
    float sq = d0 * d0 + d1 * d1 + d2 * d2 + d3 * d3 + d4 * d4 + d5 * d5 + d6 * d6 + d7 * d7;
#pragma unroll
    for (int d = 1; d < 32; d <<= 1) sq += __shfl_xor(sq, d, 32);
    const float rs = rsqrtf(sq * (1.f / 256.f) + 1e-5f);
    const float4* gp = (const float4*)g;
    const float4* bp = (const float4*)b;
    const float4 g0 = gp[lane], g1 = gp[lane + 32], b0 = bp[lane], b1 = bp[lane + 32];
    __bf16* o = out + (size_t)wave * 256;
    const int c0 = lane * 4, c1 = 128 + lane * 4;
    o[c0 + 0] = (__bf16)(d0 * rs * g0.x + b0.x);
    o[c0 + 1] = (__bf16)(d1 * rs * g0.y + b0.y);
    o[c0 + 2] = (__bf16)(d2 * rs * g0.z + b0.z);
    o[c0 + 3] = (__bf16)(d3 * rs * g0.w + b0.w);
    o[c1 + 0] = (__bf16)(d4 * rs * g1.x + b1.x);
    o[c1 + 1] = (__bf16)(d5 * rs * g1.y + b1.y);
    o[c1 + 2] = (__bf16)(d6 * rs * g1.z + b1.z);
    o[c1 + 3] = (__bf16)(d7 * rs * g1.w + b1.w);
}

// ---------------------------------------------------------------- dense GEMM
// C[M,N] = A[M,K] @ Bt[N,K]^T + bias, optional ReLU / residual add,
// bf16 and/or f32 output. One wave per 16x16 tile, v_wmma_f32_16x16x32_bf16.
__global__ void gemm_bf16_wmma_kernel(const __bf16* __restrict__ A, const __bf16* __restrict__ Bt,
                                      const float* __restrict__ bias, const float* __restrict__ residual,
                                      __bf16* __restrict__ outB, float* __restrict__ outF,
                                      int M, int N, int K, int doRelu) {
    const int lane = threadIdx.x & 31;
    const int wave = (blockIdx.x * blockDim.x + threadIdx.x) >> 5;
    const int tilesN = N >> 4;
    const int tm = wave / tilesN;
    const int tn = wave % tilesN;
    if (tm * 16 >= M) return;  // wave-uniform
    v8f acc = zero8();
    for (int k0 = 0; k0 < K; k0 += 32) {
        v16bf a = load_frag(A, K, tm * 16, k0, lane);
        v16bf b = load_frag(Bt, K, tn * 16, k0, lane);
        acc = __builtin_amdgcn_wmma_f32_16x16x32_bf16(false, a, false, b, (short)0, acc, false, false);
    }
    const int half = lane >> 4;
    const int col = tn * 16 + (lane & 15);
    const float bv = bias ? bias[col] : 0.f;
#pragma unroll
    for (int r = 0; r < 8; r++) {
        const int row = tm * 16 + r + 8 * half;
        float v = acc[r] + bv;
        if (doRelu) v = fmaxf(v, 0.f);
        if (residual) v += residual[(size_t)row * N + col];
        if (outF) outF[(size_t)row * N + col] = v;
        if (outB) outB[(size_t)row * N + col] = (__bf16)v;
    }
}

// ---------------------------------------------------------------- flash attention
// One wave per 16-row Q tile. Streams 32-column K/V tiles; fuses the
// (bias @ Wb + bb + mask) epilogue and online softmax; accumulates O in
// registers (16 v8f). bias/mask are each read exactly once and are the
// memory-bound term (320 MB total): load them NON-TEMPORAL so the 192 MB
// L2 keeps kh/vhT/qh (the heavily reused operands) resident while the
// single-use stream passes through at HBM rate.
__global__ void attn_flash_kernel(const __bf16* __restrict__ qh, const __bf16* __restrict__ kh,
                                  const __bf16* __restrict__ vhT,
                                  const float* __restrict__ bias4, const float* __restrict__ mask,
                                  const float* __restrict__ Wb, const float* __restrict__ bbp,
                                  __bf16* __restrict__ av) {
    constexpr int S = 4096;
    constexpr int D = 256;
    __shared__ __bf16 ldsQ[4][16][D];   // per-wave Q tile
    __shared__ __bf16 ldsP[4][16][32];  // per-wave P tile (C-layout -> A-layout transpose)
    const int wavein = threadIdx.x >> 5;
    const int lane = threadIdx.x & 31;
    const int tile = blockIdx.x * 4 + wavein;
    const int row0 = tile * 16;
    const int half = lane >> 4;
    const int n = lane & 15;

    {   // stage the Q tile into LDS (16 x 256 bf16, 16B chunks)
        const v8bf* src = (const v8bf*)(qh + (size_t)row0 * D);
        v8bf* dst = (v8bf*)&ldsQ[wavein][0][0];
        for (int c = lane; c < 16 * (D / 8); c += 32) dst[c] = src[c];
    }
    asm volatile("s_wait_dscnt 0" ::: "memory");

    const float wb0 = Wb[0], wb1 = Wb[1], wb2 = Wb[2], wb3 = Wb[3], bb0 = bbp[0];
    v8f o[16];
#pragma unroll
    for (int t = 0; t < 16; t++) o[t] = zero8();
    float Mr[8], Lr[8];
#pragma unroll
    for (int r = 0; r < 8; r++) { Mr[r] = -3.0e38f; Lr[r] = 0.f; }

    for (int j0 = 0; j0 < S; j0 += 32) {
        // S tile [16 x 32] = Q @ K^T over K=256
        v8f s0 = zero8(), s1 = zero8();
#pragma unroll
        for (int kk = 0; kk < 8; kk++) {
            v16bf a  = load_frag(&ldsQ[wavein][0][0], D, 0, kk * 32, lane);
            v16bf b0 = load_frag(kh, D, j0,      kk * 32, lane);
            v16bf b1 = load_frag(kh, D, j0 + 16, kk * 32, lane);
            s0 = __builtin_amdgcn_wmma_f32_16x16x32_bf16(false, a, false, b0, (short)0, s0, false, false);
            s1 = __builtin_amdgcn_wmma_f32_16x16x32_bf16(false, a, false, b1, (short)0, s1, false, false);
        }
        // epilogue: scale 1/sqrt(256), pairwise-bias projection, mask; online softmax
        float alpha[8];
#pragma unroll
        for (int r = 0; r < 8; r++) {
            const int row = row0 + r + 8 * half;
            const size_t i0 = (size_t)row * S + j0 + n;
            const fvec4 c0 = __builtin_nontemporal_load((const fvec4*)(bias4 + i0 * 4));
            const fvec4 c1 = __builtin_nontemporal_load((const fvec4*)(bias4 + (i0 + 16) * 4));
            const float m0 = __builtin_nontemporal_load(mask + i0);
            const float m1 = __builtin_nontemporal_load(mask + i0 + 16);
            float t0 = s0[r] * 0.0625f + (c0[0] * wb0 + c0[1] * wb1 + c0[2] * wb2 + c0[3] * wb3) + bb0 + m0;
            float t1 = s1[r] * 0.0625f + (c1[0] * wb0 + c1[1] * wb1 + c1[2] * wb2 + c1[3] * wb3) + bb0 + m1;
            float mx = fmaxf(t0, t1);
#pragma unroll
            for (int d = 1; d < 16; d <<= 1) mx = fmaxf(mx, __shfl_xor(mx, d, 32));
            const float mnew = fmaxf(Mr[r], mx);
            const float a_ = __expf(Mr[r] - mnew);
            const float p0 = __expf(t0 - mnew);
            const float p1 = __expf(t1 - mnew);
            float rs = p0 + p1;
#pragma unroll
            for (int d = 1; d < 16; d <<= 1) rs += __shfl_xor(rs, d, 32);
            Mr[r] = mnew;
            Lr[r] = Lr[r] * a_ + rs;
            alpha[r] = a_;
            ldsP[wavein][r + 8 * half][n]      = (__bf16)p0;
            ldsP[wavein][r + 8 * half][n + 16] = (__bf16)p1;
        }
#pragma unroll
        for (int t = 0; t < 16; t++)
#pragma unroll
            for (int r = 0; r < 8; r++) o[t][r] *= alpha[r];
        asm volatile("s_wait_dscnt 0" ::: "memory");
        // O += P @ V  (P as A-fragment from LDS, V via pre-transposed vhT)
        const v16bf pa = load_frag(&ldsP[wavein][0][0], 32, 0, 0, lane);
#pragma unroll
        for (int t = 0; t < 16; t++) {
            v16bf bv = load_frag(vhT, S, t * 16, j0, lane);
            o[t] = __builtin_amdgcn_wmma_f32_16x16x32_bf16(false, pa, false, bv, (short)0, o[t], false, false);
        }
    }
#pragma unroll
    for (int r = 0; r < 8; r++) Lr[r] = 1.f / Lr[r];
#pragma unroll
    for (int t = 0; t < 16; t++)
#pragma unroll
        for (int r = 0; r < 8; r++) {
            const int row = row0 + r + 8 * half;
            av[(size_t)row * D + t * 16 + n] = (__bf16)(o[t][r] * Lr[r]);
        }
}

// ---------------------------------------------------------------- launch
extern "C" void kernel_launch(void* const* d_in, const int* in_sizes, int n_in,
                              void* d_out, int out_size, void* d_ws, size_t ws_size,
                              hipStream_t stream) {
    (void)in_sizes; (void)n_in; (void)out_size; (void)ws_size;
    constexpr int S = 4096, D = 256, F = 1024;

    const float* q     = (const float*)d_in[0];
    const float* k     = (const float*)d_in[1];
    const float* v     = (const float*)d_in[2];
    const float* mask  = (const float*)d_in[3];
    const float* bias  = (const float*)d_in[4];
    const float* Wq    = (const float*)d_in[5];
    const float* bq    = (const float*)d_in[6];
    const float* Wk    = (const float*)d_in[7];
    const float* bk    = (const float*)d_in[8];
    const float* Wv    = (const float*)d_in[9];
    const float* bv    = (const float*)d_in[10];
    const float* Wb    = (const float*)d_in[11];
    const float* bb    = (const float*)d_in[12];
    const float* Wo    = (const float*)d_in[13];
    const float* bo    = (const float*)d_in[14];
    const float* ln1_g = (const float*)d_in[15];
    const float* ln1_b = (const float*)d_in[16];
    const float* ln2_g = (const float*)d_in[17];
    const float* ln2_b = (const float*)d_in[18];
    const float* W1    = (const float*)d_in[19];
    const float* b1    = (const float*)d_in[20];
    const float* W2    = (const float*)d_in[21];
    const float* b2    = (const float*)d_in[22];
    float* outx = (float*)d_out;

    size_t off = 0;
    auto take = [&](size_t bytes) -> void* {
        void* p = (char*)d_ws + off;
        off += (bytes + 255) & ~(size_t)255;
        return p;
    };
    __bf16* qn  = (__bf16*)take((size_t)S * D * 2);
    __bf16* kn  = (__bf16*)take((size_t)S * D * 2);
    __bf16* vn  = (__bf16*)take((size_t)S * D * 2);
    __bf16* qhb = (__bf16*)take((size_t)S * D * 2);
    __bf16* khb = (__bf16*)take((size_t)S * D * 2);
    __bf16* vhb = (__bf16*)take((size_t)S * D * 2);
    __bf16* vhT = (__bf16*)take((size_t)S * D * 2);
    __bf16* avb = (__bf16*)take((size_t)S * D * 2);
    float*  x   = (float*)take((size_t)S * D * 4);
    __bf16* xn2 = (__bf16*)take((size_t)S * D * 2);
    __bf16* h   = (__bf16*)take((size_t)S * F * 2);
    __bf16* WqT = (__bf16*)take((size_t)D * D * 2);
    __bf16* WkT = (__bf16*)take((size_t)D * D * 2);
    __bf16* WvT = (__bf16*)take((size_t)D * D * 2);
    __bf16* WoT = (__bf16*)take((size_t)D * D * 2);
    __bf16* W1T = (__bf16*)take((size_t)D * F * 2);
    __bf16* W2T = (__bf16*)take((size_t)F * D * 2);

    // weight prep (transpose + bf16)
    transpose_f32_to_bf16_kernel<<<(D * D + 255) / 256, 256, 0, stream>>>(Wq, WqT, D, D);
    transpose_f32_to_bf16_kernel<<<(D * D + 255) / 256, 256, 0, stream>>>(Wk, WkT, D, D);
    transpose_f32_to_bf16_kernel<<<(D * D + 255) / 256, 256, 0, stream>>>(Wv, WvT, D, D);
    transpose_f32_to_bf16_kernel<<<(D * D + 255) / 256, 256, 0, stream>>>(Wo, WoT, D, D);
    transpose_f32_to_bf16_kernel<<<(D * F + 255) / 256, 256, 0, stream>>>(W1, W1T, D, F);
    transpose_f32_to_bf16_kernel<<<(F * D + 255) / 256, 256, 0, stream>>>(W2, W2T, F, D);

    // LN1 on q, k, v
    layernorm_bf16_kernel<<<S / 8, 256, 0, stream>>>(q, ln1_g, ln1_b, qn, S);
    layernorm_bf16_kernel<<<S / 8, 256, 0, stream>>>(k, ln1_g, ln1_b, kn, S);
    layernorm_bf16_kernel<<<S / 8, 256, 0, stream>>>(v, ln1_g, ln1_b, vn, S);

    // Q/K/V projections
    const int projBlocks = (S / 16) * (D / 16) / 4;  // 1024 blocks of 4 waves
    gemm_bf16_wmma_kernel<<<projBlocks, 128, 0, stream>>>(qn, WqT, bq, nullptr, qhb, nullptr, S, D, D, 0);
    gemm_bf16_wmma_kernel<<<projBlocks, 128, 0, stream>>>(kn, WkT, bk, nullptr, khb, nullptr, S, D, D, 0);
    gemm_bf16_wmma_kernel<<<projBlocks, 128, 0, stream>>>(vn, WvT, bv, nullptr, vhb, nullptr, S, D, D, 0);

    // vh -> vhT for P@V B-fragments
    transpose_bf16_kernel<<<(S / 32) * (D / 32), dim3(32, 8), 0, stream>>>(vhb, vhT, S, D);

    // fused attention (bias/mask/softmax fused, flash-style)
    attn_flash_kernel<<<(S / 16) / 4, 128, 0, stream>>>(qhb, khb, vhT, bias, mask, Wb, bb, avb);

    // out-proj + residual 1 -> x (f32)
    gemm_bf16_wmma_kernel<<<projBlocks, 128, 0, stream>>>(avb, WoT, bo, q, nullptr, x, S, D, D, 0);

    // LN2
    layernorm_bf16_kernel<<<S / 8, 256, 0, stream>>>(x, ln2_g, ln2_b, xn2, S);

    // FFN
    gemm_bf16_wmma_kernel<<<(S / 16) * (F / 16) / 4, 128, 0, stream>>>(xn2, W1T, b1, nullptr, h, nullptr, S, F, D, 1);
    gemm_bf16_wmma_kernel<<<projBlocks, 128, 0, stream>>>(h, W2T, b2, x, nullptr, outx, S, D, F, 0);
}
